// LossFunction_35639638622328
// MI455X (gfx1250) — matrix operand order repaired and checked by reading, not compiled, not run on hardware
//
#include <hip/hip_runtime.h>

typedef __attribute__((ext_vector_type(16))) _Float16 v16h;
typedef __attribute__((ext_vector_type(8)))  _Float16 v8h;
typedef __attribute__((ext_vector_type(8)))  float    v8f;
typedef __attribute__((ext_vector_type(4)))  unsigned int u32x4;
typedef __attribute__((ext_vector_type(8)))  int      i32x8;
typedef __attribute__((ext_vector_type(4)))  int      i32x4;

#define NROWS   8192
#define DIM     512
#define EPSF    1e-6f
#define MARGINF 0.1f
#define NORMEPS 1e-12f

#define RPB     128                 // rows per block in argmin kernel
#define CPC     512                 // columns per chunk
#define NCHUNK  (NROWS / CPC)       // 16
#define LDA     520                 // padded LDS row stride (halves) == 1024B + 16B pad

// ---------------- TDM helper: 2D tile (rows x 512 halves) global -> LDS,
// hardware row padding of 16B per 1024B row (pad_interval=7 -> 256 DWORDs,
// pad_amount=3 -> 4 DWORDs), giving LDS row stride LDA=520 halves.
static __device__ inline void tdm_load_2d(unsigned ldsAddr, unsigned long long gaddr,
                                          unsigned tileRows) {
    u32x4 g0;
    g0[0] = 1u;                                            // count=1, user mode
    g0[1] = ldsAddr;                                       // lds_addr[31:0]
    g0[2] = (unsigned)(gaddr & 0xFFFFFFFFull);             // global_addr[31:0]
    g0[3] = (unsigned)((gaddr >> 32) & 0x1FFFFFFull)       // global_addr[56:32]
          | (2u << 30);                                    // type=2 ("image")
    i32x8 g1;
    g1[0] = (1 << 16)                                      // data_size = 2B
          | (1 << 20)                                      // pad_enable
          | (7 << 22)                                      // pad_interval: 256 DWORDs
          | (3 << 25);                                     // pad_amount: 4 DWORDs
    g1[1] = (int)(512u << 16);                             // tensor_dim0[15:0]=512
    g1[2] = (int)(tileRows << 16);                         // tensor_dim0[31:16]=0 | tensor_dim1[15:0]
    g1[3] = (int)(512u << 16);                             // tensor_dim1[31:16]=0 | tile_dim0=512
    g1[4] = (int)tileRows;                                 // tile_dim1 | tile_dim2=0
    g1[5] = 512;                                           // tensor_dim0_stride[31:0]
    g1[6] = 0;                                             // stride0[47:32]=0 | stride1 lo
    g1[7] = 0;
    i32x4 z4 = {0, 0, 0, 0};
    i32x8 z8 = {0, 0, 0, 0, 0, 0, 0, 0};
    __builtin_amdgcn_tensor_load_to_lds(g0, g1, z4, z4, z8, 0);
}

// ---------------- fragment loaders (ISA 7.12.2 f16 16x16x32 layouts)
static __device__ inline v16h ldfragA(const _Float16* base) {
    v8h lo = *(const v8h*)base;          // K = kb+0..7
    v8h hi = *(const v8h*)(base + 16);   // K = kb+16..23
    return __builtin_shufflevector(lo, hi, 0,1,2,3,4,5,6,7,8,9,10,11,12,13,14,15);
}
static __device__ inline v16h ldfragB(const _Float16* base) {
    v8h lo = *(const v8h*)base;          // K = kb+0..7
    v8h hi = *(const v8h*)(base + 16);   // K = kb+16..23
    return __builtin_shufflevector(lo, hi, 0,1,2,3,4,5,6,7,8,9,10,11,12,13,14,15);
}

// ---------------- Kernel 1: normalize rows, emit f16 copies + per-row scalars
__global__ void k1_normalize(const float* __restrict__ x,
                             _Float16* __restrict__ aH, _Float16* __restrict__ pH,
                             float* __restrict__ invA, float* __restrict__ invP,
                             float* __restrict__ cj) {
    __shared__ float s0[256], s1[256], s2[256];
    const int i = blockIdx.x;
    const int t = threadIdx.x;
    const float* xr = x + (size_t)i * (2 * DIM);
    float a0 = xr[t], a1 = xr[t + 256];
    float p0 = xr[DIM + t], p1 = xr[DIM + t + 256];
    s0[t] = a0 * a0 + a1 * a1;
    s1[t] = p0 * p0 + p1 * p1;
    s2[t] = p0 + p1;
    __syncthreads();
    for (int off = 128; off > 0; off >>= 1) {
        if (t < off) { s0[t] += s0[t + off]; s1[t] += s1[t + off]; s2[t] += s2[t + off]; }
        __syncthreads();
    }
    const float sqa = s0[0], sqp = s1[0], sp = s2[0];
    const float ia = 1.0f / fmaxf(sqrtf(sqa), NORMEPS);
    const float ip = 1.0f / fmaxf(sqrtf(sqp), NORMEPS);
    aH[(size_t)i * DIM + t]       = (_Float16)(a0 * ia);
    aH[(size_t)i * DIM + t + 256] = (_Float16)(a1 * ia);
    pH[(size_t)i * DIM + t]       = (_Float16)(p0 * ip);
    pH[(size_t)i * DIM + t + 256] = (_Float16)(p1 * ip);
    if (t == 0) {
        invA[i] = ia; invP[i] = ip;
        const float pn  = sqp * ip * ip;
        const float spn = sp * ip;
        cj[i] = pn - 2.0f * EPSF * spn;   // only column-dependent part of d2
    }
}

// ---------------- Kernel 2: TDM-fed WMMA GEMM tile + running per-row argmin
__global__ void k2_argmin(const _Float16* __restrict__ aH, const _Float16* __restrict__ pH,
                          const float* __restrict__ cj,
                          float* __restrict__ pval, int* __restrict__ pidx) {
    extern __shared__ _Float16 smem[];
    _Float16* sA  = smem;                         // RPB x LDA halves
    _Float16* sB0 = smem + RPB * LDA;             // 16 x LDA halves (buffer 0)
    _Float16* sB1 = sB0 + 16 * LDA;               // 16 x LDA halves (buffer 1)
    float*    sC  = (float*)(smem + (RPB + 32) * LDA);  // CPC floats

    const int tid  = threadIdx.x;
    const int w    = tid >> 5;
    const int l    = tid & 31;
    const int half = l >> 4;
    const int n    = l & 15;
    const int row0 = blockIdx.x * RPB;
    const int col0 = blockIdx.y * CPC;
    const int cc   = blockIdx.y;

    const unsigned sAaddr  = (unsigned)(size_t)(void*)sA;
    const unsigned sB0addr = (unsigned)(size_t)(void*)sB0;
    const unsigned sB1addr = (unsigned)(size_t)(void*)sB1;
    const unsigned long long aG = (unsigned long long)(size_t)aH;
    const unsigned long long pG = (unsigned long long)(size_t)pH;

    // Prologue: wave 0 launches TDM for the A tile and the first B tile.
    if (tid < 32) {
        tdm_load_2d(sAaddr,  aG + (unsigned long long)row0 * (DIM * 2), RPB);
        tdm_load_2d(sB0addr, pG + (unsigned long long)col0 * (DIM * 2), 16);
    }
    // stage this chunk's cj values (plain loads; covered by first barrier)
    sC[tid]       = cj[col0 + tid];
    sC[tid + 256] = cj[col0 + tid + 256];

    float minv[8];
    int   mini[8];
    #pragma unroll
    for (int r = 0; r < 8; ++r) { minv[r] = __builtin_inff(); mini[r] = 0; }

    const int rowA = 16 * w + n;

    for (int jt = 0; jt < 32; ++jt) {
        __syncthreads();   // all waves done reading the buffer we are about to overwrite
        if (tid < 32) {
            if (jt < 31) {
                // prefetch next B tile into the other buffer, then wait for the
                // CURRENT tile (TDM ops from one wave complete in order).
                tdm_load_2d((jt & 1) ? sB0addr : sB1addr,
                            pG + (unsigned long long)(col0 + 16 * (jt + 1)) * (DIM * 2), 16);
                __builtin_amdgcn_s_wait_tensorcnt(1);
            } else {
                __builtin_amdgcn_s_wait_tensorcnt(0);
            }
        }
        __syncthreads();   // broadcast completion of current tile to all waves

        const _Float16* sB = (jt & 1) ? sB1 : sB0;
        const _Float16* aBase = sA + rowA * LDA + 8 * half;
        const _Float16* bBase = sB + n * LDA + 8 * half;

        // software-pipelined K loop: load kt+1 fragments before WMMA kt
        v16h afc = ldfragA(aBase);
        v16h bfc = ldfragB(bBase);
        v8f acc = {};
        #pragma unroll
        for (int kt = 0; kt < 16; ++kt) {
            v16h afn = afc, bfn = bfc;
            if (kt < 15) {
                afn = ldfragA(aBase + 32 * (kt + 1));
                bfn = ldfragB(bBase + 32 * (kt + 1));
            }
            acc = __builtin_amdgcn_wmma_f32_16x16x32_f16(
                      false, afc, false, bfc, (short)0, acc, false, false);
            afc = afn; bfc = bfn;
        }

        const int   jg  = col0 + 16 * jt + n;
        const float cjv = sC[16 * jt + n];
        #pragma unroll
        for (int r = 0; r < 8; ++r) {
            const int m = row0 + 16 * w + r + 8 * half;
            float v = cjv - 2.0f * acc[r];
            if (jg == m) v = __builtin_inff();           // mask diagonal
            if (v < minv[r]) { minv[r] = v; mini[r] = jg; }
        }
    }

    // reduce across the 16 lanes holding the same rows
    #pragma unroll
    for (int off = 8; off >= 1; off >>= 1) {
        #pragma unroll
        for (int r = 0; r < 8; ++r) {
            float ov = __shfl_xor(minv[r], off, 32);
            int   oi = __shfl_xor(mini[r], off, 32);
            if (ov < minv[r] || (ov == minv[r] && oi < mini[r])) {
                minv[r] = ov; mini[r] = oi;
            }
        }
    }
    if (n == 0) {
        #pragma unroll
        for (int r = 0; r < 8; ++r) {
            const int grow = row0 + 16 * w + r + 8 * half;
            pval[(size_t)grow * NCHUNK + cc] = minv[r];
            pidx[(size_t)grow * NCHUNK + cc] = mini[r];
        }
    }
}

// ---------------- Kernel 2b: combine chunk partials -> negidx
__global__ void k2b_reduce(const float* __restrict__ pval, const int* __restrict__ pidx,
                           int* __restrict__ negidx) {
    const int i = blockIdx.x * blockDim.x + threadIdx.x;
    if (i >= NROWS) return;
    float best = __builtin_inff();
    int   bi = 0;
    #pragma unroll
    for (int c = 0; c < NCHUNK; ++c) {
        float v = pval[(size_t)i * NCHUNK + c];
        if (v < best) { best = v; bi = pidx[(size_t)i * NCHUNK + c]; }
    }
    negidx[i] = bi;
}

// ---------------- Kernel 3: per-row pos/neg distances (full f32) -> relu loss
__global__ void k3_loss(const float* __restrict__ x, const float* __restrict__ invA,
                        const float* __restrict__ invP, const int* __restrict__ negidx,
                        float* __restrict__ rowloss) {
    __shared__ float s0[256], s1[256];
    const int i = blockIdx.x, t = threadIdx.x;
    const int j = negidx[i];
    const float ia = invA[i], ip = invP[i], ipj = invP[j];
    const float* xi = x + (size_t)i * (2 * DIM);
    const float* xj = x + (size_t)j * (2 * DIM);
    float pos = 0.f, neg = 0.f;
    #pragma unroll
    for (int u = 0; u < 2; ++u) {
        const int k = t + 256 * u;
        const float a  = xi[k] * ia;
        const float pi = xi[DIM + k] * ip;
        const float pj = xj[DIM + k] * ipj;
        const float dp = a - pi + EPSF;
        const float dn = a - pj + EPSF;
        pos += dp * dp;
        neg += dn * dn;
    }
    s0[t] = pos; s1[t] = neg;
    __syncthreads();
    for (int off = 128; off > 0; off >>= 1) {
        if (t < off) { s0[t] += s0[t + off]; s1[t] += s1[t + off]; }
        __syncthreads();
    }
    if (t == 0) rowloss[i] = fmaxf(s0[0] - s1[0] + MARGINF, 0.0f);
}

// ---------------- Kernel 4: deterministic mean
__global__ void k4_mean(const float* __restrict__ rowloss, float* __restrict__ out) {
    __shared__ float s[256];
    const int t = threadIdx.x;
    float acc = 0.f;
    for (int k = t; k < NROWS; k += 256) acc += rowloss[k];
    s[t] = acc;
    __syncthreads();
    for (int off = 128; off > 0; off >>= 1) {
        if (t < off) s[t] += s[t + off];
        __syncthreads();
    }
    if (t == 0) out[0] = s[0] / (float)NROWS;
}

extern "C" void kernel_launch(void* const* d_in, const int* in_sizes, int n_in,
                              void* d_out, int out_size, void* d_ws, size_t ws_size,
                              hipStream_t stream) {
    const float* x = (const float*)d_in[0];
    char* ws = (char*)d_ws;
    _Float16* aH      = (_Float16*)(ws);                    //  8 MB
    _Float16* pH      = (_Float16*)(ws + 8388608);          //  8 MB
    float*    invA    = (float*)(ws + 16777216);            // 32 KB
    float*    invP    = (float*)(ws + 16809984);            // 32 KB
    float*    cj      = (float*)(ws + 16842752);            // 32 KB
    float*    pval    = (float*)(ws + 16875520);            // 512 KB
    int*      pidx    = (int*)  (ws + 17399808);            // 512 KB
    int*      negidx  = (int*)  (ws + 17924096);            // 32 KB
    float*    rowloss = (float*)(ws + 17956864);            // 32 KB
    float*    out     = (float*)d_out;

    k1_normalize<<<NROWS, 256, 0, stream>>>(x, aH, pH, invA, invP, cj);

    const size_t smemBytes = (size_t)(RPB + 32) * LDA * sizeof(_Float16)
                           + (size_t)CPC * sizeof(float);   // ~168.4 KB
    k2_argmin<<<dim3(NROWS / RPB, NCHUNK), 256, smemBytes, stream>>>(aH, pH, cj, pval, pidx);

    k2b_reduce<<<NROWS / 256, 256, 0, stream>>>(pval, pidx, negidx);
    k3_loss<<<NROWS, 256, 0, stream>>>(x, invA, invP, negidx, rowloss);
    k4_mean<<<1, 256, 0, stream>>>(rowloss, out);
}
